// GCN_29798483100071
// MI455X (gfx1250) — compile-verified, matrix operand order checked by reading, not run on hardware
//
#include <hip/hip_runtime.h>

typedef __attribute__((ext_vector_type(16))) _Float16 v16h;
typedef __attribute__((ext_vector_type(8)))  _Float16 v8h;
typedef __attribute__((ext_vector_type(8)))  float    v8f;

// ---------------------------------------------------------------------------
// Degree / normalization
// ---------------------------------------------------------------------------
__global__ void fill_f32(float* __restrict__ p, float v, int n) {
    int i = blockIdx.x * blockDim.x + threadIdx.x;
    if (i < n) p[i] = v;
}

__global__ void edge_deg(const int* __restrict__ dst, float* __restrict__ deg, int E) {
    int e = blockIdx.x * blockDim.x + threadIdx.x;
    if (e < E) atomicAdd(&deg[dst[e]], 1.0f);
}

__global__ void deg_rsqrt(const float* __restrict__ deg, float* __restrict__ dinv, int n) {
    int i = blockIdx.x * blockDim.x + threadIdx.x;
    if (i < n) dinv[i] = __frsqrt_rn(deg[i]);   // deg >= 1 always (self-loop)
}

// ---------------------------------------------------------------------------
// f32 -> f16 casts (feature matrix, and weight matrix transposed to [N][K])
// ---------------------------------------------------------------------------
__global__ void cast_f32_f16(const float* __restrict__ in, _Float16* __restrict__ out, int n) {
    int i = blockIdx.x * blockDim.x + threadIdx.x;
    if (i < n) out[i] = (_Float16)in[i];
}

template <int K, int NC>
__global__ void castT_f32_f16(const float* __restrict__ W, _Float16* __restrict__ Wt) {
    int i = blockIdx.x * blockDim.x + threadIdx.x;   // over K*NC
    if (i >= K * NC) return;
    int k = i / NC, n = i % NC;
    Wt[n * K + k] = (_Float16)W[i];                  // Wt is [NC][K], contiguous in K
}

// ---------------------------------------------------------------------------
// GEMM: C[N x DOUT_] = A[N x K] * W[K x DOUT_]   (W given transposed: Bt[DOUT_][K])
// One 16x16 output tile per wave32; v_wmma_f32_16x16x32_f16 along K.
// ---------------------------------------------------------------------------
template <int K, int DOUT_>
__global__ __launch_bounds__(256) void gcn_gemm(const _Float16* __restrict__ A,
                                                const _Float16* __restrict__ Bt,
                                                float* __restrict__ C, int nRows) {
    const int lane = threadIdx.x & 31;
    const int wave = threadIdx.x >> 5;
    const int tile = blockIdx.x * (blockDim.x >> 5) + wave;
    constexpr int NT = DOUT_ / 16;
    const int totalTiles = (nRows / 16) * NT;
    if (tile >= totalTiles) return;                  // wave-uniform guard (EXEC all-ones)

    const int m0 = (tile / NT) * 16;
    const int n0 = (tile % NT) * 16;
    const int half_id = lane >> 4;                   // K-group select per ISA A/B layout
    const int mr = lane & 15;                        // M for A-frag, N for B-frag

    const _Float16* arow = A  + (size_t)(m0 + mr) * K + half_id * 8;
    const _Float16* brow = Bt + (size_t)(n0 + mr) * K + half_id * 8;

    v8f acc = {};
#pragma unroll
    for (int k0 = 0; k0 < K; k0 += 32) {
        v8h alo = *(const v8h*)(arow + k0);          // K {0..7} or {8..15}  (+k0)
        v8h ahi = *(const v8h*)(arow + k0 + 16);     // K {16..23} or {24..31}
        v8h blo = *(const v8h*)(brow + k0);
        v8h bhi = *(const v8h*)(brow + k0 + 16);
        v16h a = __builtin_shufflevector(alo, ahi, 0,1,2,3,4,5,6,7,8,9,10,11,12,13,14,15);
        v16h b = __builtin_shufflevector(blo, bhi, 0,1,2,3,4,5,6,7,8,9,10,11,12,13,14,15);
        acc = __builtin_amdgcn_wmma_f32_16x16x32_f16(false, a, false, b,
                                                     (short)0, acc, false, false);
    }

    // D layout: lane = N (mr), VGPR r -> row (r + 8*half_id)
    float* crow = C + (size_t)m0 * DOUT_ + n0;
#pragma unroll
    for (int r = 0; r < 8; r++)
        crow[(size_t)(r + 8 * half_id) * DOUT_ + mr] = acc[r];
}

// ---------------------------------------------------------------------------
// Edge scatter: agg[dst] += dinv[src]*dinv[dst] * xw[src]
// D/4 threads per edge, float4 per thread; L2-resident random access.
// ---------------------------------------------------------------------------
template <int D>
__global__ __launch_bounds__(256) void gcn_scatter(const float* __restrict__ xw,
                                                   const int* __restrict__ src,
                                                   const int* __restrict__ dst,
                                                   const float* __restrict__ dinv,
                                                   float* __restrict__ agg, int E) {
    constexpr int TPE = D / 4;
    long long t = (long long)blockIdx.x * blockDim.x + threadIdx.x;
    int e = (int)(t / TPE);
    int lane = (int)(t % TPE);
    if (e >= E) return;
    int s = src[e], d = dst[e];
    float w = dinv[s] * dinv[d];
    const float4 v = *(const float4*)(xw + (size_t)s * D + lane * 4);
    float* out = agg + (size_t)d * D + lane * 4;
    atomicAdd(out + 0, v.x * w);
    atomicAdd(out + 1, v.y * w);
    atomicAdd(out + 2, v.z * w);
    atomicAdd(out + 3, v.w * w);
}

// ---------------------------------------------------------------------------
// Finalize: out = (agg + dinv^2 * xw) / cnt + bias  [+ relu]; emit f16 (next
// layer GEMM input) or f32 (final output).
// ---------------------------------------------------------------------------
template <int D, bool RELU>
__global__ void gcn_finalize_f16(const float* __restrict__ agg, const float* __restrict__ xw,
                                 const float* __restrict__ dinv, const float* __restrict__ cnt,
                                 const float* __restrict__ bias, _Float16* __restrict__ out,
                                 int total) {
    int i = blockIdx.x * blockDim.x + threadIdx.x;
    if (i >= total) return;
    int node = i / D, f = i % D;
    float di = dinv[node];
    float v = (agg[i] + di * di * xw[i]) / cnt[node] + bias[f];
    if (RELU) v = fmaxf(v, 0.0f);
    out[i] = (_Float16)v;
}

template <int D>
__global__ void gcn_finalize_f32(const float* __restrict__ agg, const float* __restrict__ xw,
                                 const float* __restrict__ dinv, const float* __restrict__ cnt,
                                 const float* __restrict__ bias, float* __restrict__ out,
                                 int total) {
    int i = blockIdx.x * blockDim.x + threadIdx.x;
    if (i >= total) return;
    int node = i / D, f = i % D;
    float di = dinv[node];
    out[i] = (agg[i] + di * di * xw[i]) / cnt[node] + bias[f];
}

// ---------------------------------------------------------------------------
static inline int cdiv(long long a, int b) { return (int)((a + b - 1) / b); }

extern "C" void kernel_launch(void* const* d_in, const int* in_sizes, int n_in,
                              void* d_out, int out_size, void* d_ws, size_t ws_size,
                              hipStream_t stream) {
    constexpr int DIN = 128, DH = 128, DO = 64;
    const float* x  = (const float*)d_in[0];
    const int*   ei = (const int*)d_in[1];
    const float* W1 = (const float*)d_in[2];
    const float* b1 = (const float*)d_in[3];
    const float* W2 = (const float*)d_in[4];
    const float* b2 = (const float*)d_in[5];
    const float* W3 = (const float*)d_in[6];
    const float* b3 = (const float*)d_in[7];
    float* out = (float*)d_out;

    const int N = in_sizes[0] / DIN;     // 100000
    const int E = in_sizes[1] / 2;       // 1600000
    const int* src = ei;
    const int* dst = ei + E;

    // ---- workspace carve (all 256B aligned) ----
    auto au = [](size_t s) { return (s + 255) & ~(size_t)255; };
    char* p = (char*)d_ws;
    float*    deg  = (float*)p;    p += au((size_t)N * 4);
    float*    dinv = (float*)p;    p += au((size_t)N * 4);
    _Float16* h16  = (_Float16*)p; p += au((size_t)N * DH * 2);
    _Float16* w16t = (_Float16*)p; p += au((size_t)DIN * DH * 2);
    float*    xw   = (float*)p;    p += au((size_t)N * DH * 4);
    float*    agg  = (float*)p;    p += au((size_t)N * DH * 4);
    (void)ws_size;

    const int B = 256;

    // ---- normalization prep ----
    fill_f32<<<cdiv(N, B), B, 0, stream>>>(deg, 1.0f, N);                 // self-loop
    edge_deg<<<cdiv(E, B), B, 0, stream>>>(dst, deg, E);
    deg_rsqrt<<<cdiv(N, B), B, 0, stream>>>(deg, dinv, N);

    // ---- layer 1: x @ W1 -> scatter -> relu -> h16 ----
    cast_f32_f16<<<cdiv((long long)N * DIN, B), B, 0, stream>>>(x, h16, N * DIN);
    castT_f32_f16<DIN, DH><<<cdiv(DIN * DH, B), B, 0, stream>>>(W1, w16t);
    gcn_gemm<DIN, DH><<<(N / 16) * (DH / 16) / 8, B, 0, stream>>>(h16, w16t, xw, N);
    hipMemsetAsync(agg, 0, (size_t)N * DH * 4, stream);
    gcn_scatter<DH><<<cdiv((long long)E * (DH / 4), B), B, 0, stream>>>(xw, src, dst, dinv, agg, E);
    gcn_finalize_f16<DH, true><<<cdiv((long long)N * DH, B), B, 0, stream>>>(
        agg, xw, dinv, deg, b1, h16, N * DH);

    // ---- layer 2 ----
    castT_f32_f16<DH, DH><<<cdiv(DH * DH, B), B, 0, stream>>>(W2, w16t);
    gcn_gemm<DH, DH><<<(N / 16) * (DH / 16) / 8, B, 0, stream>>>(h16, w16t, xw, N);
    hipMemsetAsync(agg, 0, (size_t)N * DH * 4, stream);
    gcn_scatter<DH><<<cdiv((long long)E * (DH / 4), B), B, 0, stream>>>(xw, src, dst, dinv, agg, E);
    gcn_finalize_f16<DH, true><<<cdiv((long long)N * DH, B), B, 0, stream>>>(
        agg, xw, dinv, deg, b2, h16, N * DH);

    // ---- layer 3 (no relu, f32 output) ----
    castT_f32_f16<DH, DO><<<cdiv(DH * DO, B), B, 0, stream>>>(W3, w16t);
    gcn_gemm<DH, DO><<<(N / 16) * (DO / 16) / 8, B, 0, stream>>>(h16, w16t, xw, N);
    hipMemsetAsync(agg, 0, (size_t)N * DO * 4, stream);
    gcn_scatter<DO><<<cdiv((long long)E * (DO / 4), B), B, 0, stream>>>(xw, src, dst, dinv, agg, E);
    gcn_finalize_f32<DO><<<cdiv((long long)N * DO, B), B, 0, stream>>>(
        agg, xw, dinv, deg, b3, out, N * DO);
}
